// kernel_generated_3_62904091017470
// MI455X (gfx1250) — matrix operand order repaired and checked by reading, not compiled, not run on hardware
//
#include <hip/hip_runtime.h>
#include <cstddef>

// ---------------------------------------------------------------------------
// Shift-conv as implicit GEMM on CDNA5 (gfx1250, wave32) using f32-accurate
// f16 split emulation on V_WMMA_F32_16X16X32_F16.
//
// out[b,l,n,o] = sum_{j,k,i} A[(n,o)][b][(j,k,i)] * w[j,k,i,l]
//   A[(n,o)][b][jki] = mask(0<=o+i-1<14) *
//        ( pad(x)[b,    j, n+k-1,          (o+i-2)%14]
//        + pad(x)[b, 64+j, ((n-1)%14)+k-1, (o+i-2)%14] )
//
// Precision: a = ah + al (f16 hi/lo);  a*b ~= ah*bh + ah*bl + al*bh
//   -> 3x WMMA 16x16x32 per K=32 chunk, ~1e-6 rel err (vs 8x f32 16x16x4).
//
// Packs (phase 1) are written directly in WMMA fragment lane layout so every
// GEMM operand load is a coalesced 32B-per-lane v16h from the 192MB L2:
//   A frag (16x32 MxK):  lane L: M=L%16; vgpr r, half s:
//                        K=(r/4)*16 + (L>=16?8:0) + (r%4)*2 + s
//   B frag (32x16 KxN):  lane L: N=L%16; vgpr r, half s:
//                        K=(r/4)*16 + (L>=16?8:0) + (r%4)*2 + s
//
// Workspace: A hi+lo 57.8 MB + B hi+lo 0.59 MB (L2-resident).
// ---------------------------------------------------------------------------

typedef _Float16 half_t;
typedef __attribute__((ext_vector_type(16))) _Float16 v16h;
typedef __attribute__((ext_vector_type(8)))  float    v8f;

#define NP   196   // 14*14 spatial positions
#define NMB  8     // 128 batch rows / 16
#define NCH  18    // 576 K / 32
#define APK_HALVES ((size_t)NP * NMB * NCH * 32 * 16)  // 14,450,688
#define BPK_HALVES ((size_t)16 * NCH * 32 * 16)        // 147,456

union V16U { v16h v; half_t h[16]; };

// ------------------------- Phase 1a: weight pack ---------------------------
__global__ __launch_bounds__(256) void pack_w_kernel(const float* __restrict__ w,
                                                     half_t* __restrict__ Bhi,
                                                     half_t* __restrict__ Blo) {
    const int t    = blockIdx.x * 256 + threadIdx.x;   // 16*18*32 = 9216 threads
    const int lane = t & 31;
    const int ch   = (t >> 5) % NCH;
    const int nb   = t / (NCH * 32);
    const int l    = nb * 16 + (lane & 15);
    V16U hi, lo;
#pragma unroll
    for (int r = 0; r < 8; ++r) {
#pragma unroll
        for (int s = 0; s < 2; ++s) {
            const int kloc = (r >> 2) * 16 + ((lane & 16) ? 8 : 0) + (r & 3) * 2 + s;  // B frag K map
            const int kk   = ch * 32 + kloc;
            const float v  = w[(size_t)kk * 256 + l];
            const half_t h = (half_t)v;
            hi.h[r * 2 + s] = h;
            lo.h[r * 2 + s] = (half_t)(v - (float)h);
        }
    }
    *(v16h*)(Bhi + (size_t)t * 16) = hi.v;
    *(v16h*)(Blo + (size_t)t * 16) = lo.v;
}

// ------------------------- Phase 1b: im2col pack ---------------------------
__device__ __forceinline__ float a_val(const float* __restrict__ x,
                                       int b, int n, int o, int kk) {
    const int i = kk % 3;
    const int k = (kk / 3) % 3;
    const int j = kk / 9;
    const int oi = o + i - 1;
    if ((unsigned)oi >= 14u) return 0.0f;              // width unfold mask
    int wc = oi - 1; if (wc < 0) wc += 14;             // (o+i-2) mod 14 (roll)
    float v = 0.0f;
    const int h1 = n + k - 1;                          // group 0, height pad
    if ((unsigned)h1 < 14u)
        v += x[(((size_t)b * 128 + j) * 14 + h1) * 14 + wc];
    int n2 = n - 1; if (n2 < 0) n2 = 13;               // height roll (wrap)
    const int h2 = n2 + k - 1;                         // group 1, height pad
    if ((unsigned)h2 < 14u)
        v += x[(((size_t)b * 128 + 64 + j) * 14 + h2) * 14 + wc];
    return v;
}

__global__ __launch_bounds__(256) void pack_a_kernel(const float* __restrict__ x,
                                                     half_t* __restrict__ Ahi,
                                                     half_t* __restrict__ Alo) {
    const int t    = blockIdx.x * 256 + threadIdx.x;   // 196*8*18*32 = 903,168
    const int lane = t & 31;
    const int ch   = (t >> 5) % NCH;
    const int mb   = (t / (32 * NCH)) & 7;
    const int p    = t / (32 * NCH * NMB);
    const int n = p / 14, o = p % 14;
    const int m = mb * 16 + (lane & 15);
    V16U hi, lo;
#pragma unroll
    for (int r = 0; r < 8; ++r) {
#pragma unroll
        for (int s = 0; s < 2; ++s) {
            const int kloc = (r >> 2) * 16 + ((lane & 16) ? 8 : 0)
                           + (r & 3) * 2 + s;          // A frag K map
            const float v  = a_val(x, m, n, o, ch * 32 + kloc);
            const half_t h = (half_t)v;
            hi.h[r * 2 + s] = h;
            lo.h[r * 2 + s] = (half_t)(v - (float)h);
        }
    }
    *(v16h*)(Ahi + (size_t)t * 16) = hi.v;
    *(v16h*)(Alo + (size_t)t * 16) = lo.v;
}

// ------------------------- Phase 2: WMMA GEMM ------------------------------
__global__ __launch_bounds__(256) void gemm_wmma_kernel(const half_t* __restrict__ Ahi,
                                                        const half_t* __restrict__ Alo,
                                                        const half_t* __restrict__ Bhi,
                                                        const half_t* __restrict__ Blo,
                                                        float* __restrict__ out) {
    const int p     = blockIdx.x % NP;        // spatial position (n,o)
    const int nhalf = blockIdx.x / NP;        // which 128 of 256 out-channels
    const int wv    = threadIdx.x >> 5;
    const int lane  = threadIdx.x & 31;
    const int mrow  = wv >> 1;                // 0..3
    const int ncol  = wv & 1;                 // 0..1
    const int mb0   = mrow * 2;               // wave owns M blocks mb0, mb0+1
    const int nb0   = nhalf * 8 + ncol * 4;   // wave owns N blocks nb0..nb0+3

    const size_t a0off = ((size_t)(p * NMB + mb0)     * NCH * 32 + lane) * 16;
    const size_t a1off = ((size_t)(p * NMB + mb0 + 1) * NCH * 32 + lane) * 16;
    const half_t* a0h = Ahi + a0off;
    const half_t* a0l = Alo + a0off;
    const half_t* a1h = Ahi + a1off;
    const half_t* a1l = Alo + a1off;
    const size_t b0off = ((size_t)(nb0 + 0) * NCH * 32 + lane) * 16;
    const size_t b1off = ((size_t)(nb0 + 1) * NCH * 32 + lane) * 16;
    const size_t b2off = ((size_t)(nb0 + 2) * NCH * 32 + lane) * 16;
    const size_t b3off = ((size_t)(nb0 + 3) * NCH * 32 + lane) * 16;

    v8f acc[2][4] = {};

#define WMMA16(A, B, C) \
    __builtin_amdgcn_wmma_f32_16x16x32_f16(false, (A), false, (B), (short)0, (C), false, false)

#pragma unroll 2
    for (int ch = 0; ch < NCH; ++ch) {
        const size_t off = (size_t)ch * 512;  // 32 lanes * 16 halves per chunk
        v16h A0H = *(const v16h*)(a0h + off);
        v16h A1H = *(const v16h*)(a1h + off);
        v16h A0L = *(const v16h*)(a0l + off);
        v16h A1L = *(const v16h*)(a1l + off);
        v16h B0H = *(const v16h*)(Bhi + b0off + off);
        v16h B1H = *(const v16h*)(Bhi + b1off + off);
        v16h B2H = *(const v16h*)(Bhi + b2off + off);
        v16h B3H = *(const v16h*)(Bhi + b3off + off);
        v16h B0L = *(const v16h*)(Blo + b0off + off);
        v16h B1L = *(const v16h*)(Blo + b1off + off);
        v16h B2L = *(const v16h*)(Blo + b2off + off);
        v16h B3L = *(const v16h*)(Blo + b3off + off);

        // term 1: ah*bh   (accumulators retouched every 8 WMMAs)
        acc[0][0] = WMMA16(A0H, B0H, acc[0][0]);
        acc[0][1] = WMMA16(A0H, B1H, acc[0][1]);
        acc[0][2] = WMMA16(A0H, B2H, acc[0][2]);
        acc[0][3] = WMMA16(A0H, B3H, acc[0][3]);
        acc[1][0] = WMMA16(A1H, B0H, acc[1][0]);
        acc[1][1] = WMMA16(A1H, B1H, acc[1][1]);
        acc[1][2] = WMMA16(A1H, B2H, acc[1][2]);
        acc[1][3] = WMMA16(A1H, B3H, acc[1][3]);
        // term 2: ah*bl
        acc[0][0] = WMMA16(A0H, B0L, acc[0][0]);
        acc[0][1] = WMMA16(A0H, B1L, acc[0][1]);
        acc[0][2] = WMMA16(A0H, B2L, acc[0][2]);
        acc[0][3] = WMMA16(A0H, B3L, acc[0][3]);
        acc[1][0] = WMMA16(A1H, B0L, acc[1][0]);
        acc[1][1] = WMMA16(A1H, B1L, acc[1][1]);
        acc[1][2] = WMMA16(A1H, B2L, acc[1][2]);
        acc[1][3] = WMMA16(A1H, B3L, acc[1][3]);
        // term 3: al*bh
        acc[0][0] = WMMA16(A0L, B0H, acc[0][0]);
        acc[0][1] = WMMA16(A0L, B1H, acc[0][1]);
        acc[0][2] = WMMA16(A0L, B2H, acc[0][2]);
        acc[0][3] = WMMA16(A0L, B3H, acc[0][3]);
        acc[1][0] = WMMA16(A1L, B0H, acc[1][0]);
        acc[1][1] = WMMA16(A1L, B1H, acc[1][1]);
        acc[1][2] = WMMA16(A1L, B2H, acc[1][2]);
        acc[1][3] = WMMA16(A1L, B3H, acc[1][3]);
    }
#undef WMMA16

    // C/D layout: VGPR r -> M = r (lanes 0-15) / r+8 (lanes 16-31), N = lane%16.
    const int l_lo = lane & 15;
    const int mo   = (lane >> 4) << 3;        // 0 or 8
#pragma unroll
    for (int mi = 0; mi < 2; ++mi) {
#pragma unroll
        for (int ni = 0; ni < 4; ++ni) {
            const size_t l = (size_t)(nb0 + ni) * 16 + l_lo;
#pragma unroll
            for (int r = 0; r < 8; ++r) {
                const size_t b = (size_t)(mb0 + mi) * 16 + mo + r;
                // out[b, l, n, o] ; flat = b*256*196 + l*196 + p
                out[b * 50176 + l * 196 + p] = acc[mi][ni][r];
            }
        }
    }
}

// ------------------------------ launcher -----------------------------------
extern "C" void kernel_launch(void* const* d_in, const int* in_sizes, int n_in,
                              void* d_out, int out_size, void* d_ws, size_t ws_size,
                              hipStream_t stream) {
    (void)in_sizes; (void)n_in; (void)out_size; (void)ws_size;
    const float* x = (const float*)d_in[0];   // (128,128,14,14) f32
    const float* w = (const float*)d_in[1];   // (64,3,3,256)   f32
    float* out = (float*)d_out;               // (128,256,14,14) f32

    half_t* Ahi = (half_t*)d_ws;              // 28.9 MB
    half_t* Alo = Ahi + APK_HALVES;           // 28.9 MB
    half_t* Bhi = Alo + APK_HALVES;           // 0.29 MB
    half_t* Blo = Bhi + BPK_HALVES;           // 0.29 MB

    pack_w_kernel<<<36,   256, 0, stream>>>(w, Bhi, Blo);   // 9,216 threads
    pack_a_kernel<<<3528, 256, 0, stream>>>(x, Ahi, Alo);   // 903,168 threads
    gemm_wmma_kernel<<<392, 256, 0, stream>>>(Ahi, Alo, Bhi, Blo, out);
}
